// ResidualSwinTransformerBlock_20194936226640
// MI455X (gfx1250) — matrix operand order repaired and checked by reading, not compiled, not run on hardware
//
#include <hip/hip_runtime.h>
#include <hip/hip_bf16.h>

// MI455X (gfx1250) fused Swin-block pair + 3x3 conv residual.
// bf16 WMMA (v_wmma_f32_16x16x32_bf16) for every GEMM, one window (64 tokens)
// per 256-thread workgroup. Weights pre-packed to bf16 fragment-linear layout.
// Wave mapping (M-pair x N-quarter) gives 2 WMMAs per B-fragment fetch; LDS
// row strides padded to odd word counts for bank-conflict-free fragment reads.
// Workspace: ~224 MB (3 NHWC f32 buffers + packed weights).

typedef unsigned short u16;
typedef __bf16 bf16_t;
typedef bf16_t v16bf __attribute__((ext_vector_type(16)));
typedef float  v8f   __attribute__((ext_vector_type(8)));

#define CP 198   // padded activation row stride (u16): 99 words, gcd(99,64)=1
#define SP 70    // padded softmax row stride: 35 words
#define HP 742   // padded mlp-hidden row stride: 371 words

union FragU { v16bf v; unsigned u[8]; };

static __device__ __forceinline__ u16 f2bf(float f) {
  unsigned u = __float_as_uint(f);
  unsigned r = u + 0x7fffu + ((u >> 16) & 1u);  // round-to-nearest-even
  return (u16)(r >> 16);
}

static __device__ __forceinline__ v8f wmma_bf16(v16bf a, v16bf b, v8f c) {
  return __builtin_amdgcn_wmma_f32_16x16x32_bf16(false, a, false, b, (short)0, c,
                                                 false, false);
}

static __device__ __forceinline__ v8f v8f_fill(float x) {
  v8f c;
#pragma unroll
  for (int v = 0; v < 8; ++v) c[v] = x;
  return c;
}

// A-matrix 16x32 bf16 fragment from row-major LDS (k = 16*(lane/16) + s, matching B).
static __device__ __forceinline__ v16bf frag_a_lds(const u16* base, int stride) {
  int lane = threadIdx.x & 31;
  const u16* rp = base + (lane & 15) * stride;
  int koff = (lane >> 4) << 4;
  FragU f;
#pragma unroll
  for (int v = 0; v < 8; ++v) {
    int k = koff + v * 2;
    f.u[v] = *(const unsigned*)(rp + k);
  }
  return f.v;
}

// B-matrix 32x16: element (k,n) at base[n*stride + k] (K contiguous in row).
static __device__ __forceinline__ v16bf frag_b_nmaj(const u16* base, int stride) {
  int lane = threadIdx.x & 31;
  const u16* rp = base + (lane & 15) * stride + ((lane >> 4) << 4);
  FragU f;
#pragma unroll
  for (int v = 0; v < 8; ++v) f.u[v] = *(const unsigned*)(rp + v * 2);
  return f.v;
}

// B-matrix 32x16: element (k,n) at base[k*stride + n] (K strided).
static __device__ __forceinline__ v16bf frag_b_kmaj(const u16* base, int stride) {
  int lane = threadIdx.x & 31;
  int nn = lane & 15, kb = (lane >> 4) << 4;
  FragU f;
#pragma unroll
  for (int v = 0; v < 8; ++v) {
    int k = kb + v * 2;
    unsigned lo = base[k * stride + nn];
    unsigned hi = base[(k + 1) * stride + nn];
    f.u[v] = lo | (hi << 16);
  }
  return f.v;
}

// B fragment from global, pre-packed lane-major (32 contiguous bytes / lane).
static __device__ __forceinline__ v16bf frag_b_glob(const u16* tile) {
  int lane = threadIdx.x & 31;
  const unsigned* p = (const unsigned*)(tile + lane * 16);
  FragU f;
#pragma unroll
  for (int v = 0; v < 8; ++v) f.u[v] = p[v];
  return f.v;
}

static __device__ __forceinline__ int regof(int g) {  // shift-window regions
  return (g < 32) ? 0 : ((g < 36) ? 1 : 2);           // RES-WS=32, RES-SHIFT=36
}

// ---------------- weight packing ----------------
// dst tile (nt,kt) at (nt*KT+kt)*512; halfword q of lane L is element
// B[k=(L/16)*16+q][n=L%16]; B[k][n] = W[o(n_pad)][i=kt*32+k].
__global__ void prep_weight_kernel(u16* __restrict__ dst, const float* __restrict__ src,
                                   int o_stride, int i_stride, int src_off,
                                   int out_real, int in_real, int KT, int qkv_mode,
                                   int total) {
  int e = blockIdx.x * blockDim.x + threadIdx.x;
  if (e >= total) return;
  int tile = e >> 9, r = e & 511;
  int lane = r >> 4, q = r & 15;
  int nt = tile / KT, kt = tile % KT;
  int nn = lane & 15, kb = (lane >> 4) << 4;
  int i = kt * 32 + kb + q;
  int opad = nt * 16 + nn, o;
  if (qkv_mode) {  // per-head padded qkv channel map: seg(0..2) x head(6) x 32
    int seg = opad / 192, cc = opad % 192, hh = cc >> 5, dd = cc & 31;
    o = (dd < 30) ? (seg * 180 + hh * 30 + dd) : -1;
  } else {
    o = (opad < out_real) ? opad : -1;
  }
  float val = (o >= 0 && i < in_real)
                  ? src[src_off + (long)o * o_stride + (long)i * i_stride]
                  : 0.f;
  dst[e] = f2bf(val);
}

__global__ void nchw_to_nhwc_kernel(const float* __restrict__ src,
                                    float* __restrict__ dst) {
  long i = (long)blockIdx.x * blockDim.x + threadIdx.x;
  if (i >= 18432000L) return;
  int ch = (int)(i % 180);
  long t = i / 180;
  int pix = (int)(t % 1600);
  int b = (int)(t / 1600);
  dst[i] = src[((long)b * 180 + ch) * 1600 + pix];
}

// ---------------- fused Swin block (one window per workgroup) ----------------
__global__ __launch_bounds__(256, 1) void swin_block_kernel(
    const float* __restrict__ hin, float* __restrict__ hout,
    const u16* __restrict__ wqkv, const u16* __restrict__ wproj,
    const u16* __restrict__ wfc1, const u16* __restrict__ wfc2,
    const float* __restrict__ ln1g, const float* __restrict__ ln1b,
    const float* __restrict__ qkvb, const float* __restrict__ projb,
    const float* __restrict__ rpb,
    const float* __restrict__ ln2g, const float* __restrict__ ln2b,
    const float* __restrict__ fc1b, const float* __restrict__ fc2b, int shift) {
  __shared__ u16 smem[77568];           // 151.5 KB, phase-reused
  u16* XLN = smem;                      // 64xCP bf16 (xln1, later xln2)
  u16* Qb = smem + 12672;               // 64xCP
  u16* Kb = smem + 25344;               // 64xCP (reused as attn-out)
  u16* Vb = smem + 38016;               // 64xCP
  u16* Sb = smem + 50688;               // 6 heads x 64xSP bf16 (reused as X2)
  u16* ATT = Kb;
  float* X2 = (float*)(smem + 50688);   // 64x180 f32
  u16* H1 = smem + 12672;               // 64xHP bf16 (mlp hidden)

  const int bidx = blockIdx.x;
  const int bb = bidx / 25, wid = bidx % 25;
  const int wr = wid / 5, wc = wid % 5;
  const int wave = threadIdx.x >> 5, lane = threadIdx.x & 31;
  const int nn = lane & 15, rowoff = (lane >> 4) << 3;
  const int mtp = wave & 1, ntq = wave >> 1;   // M-pair, N-quarter
  const int mt0 = mtp * 2, mt1 = mt0 + 1;

  auto tokbase = [&](int t) {  // roll(-s) then unroll(+s) == identity on coords
    int gr = (wr * 8 + (t >> 3) + shift) % 40;
    int gc = (wc * 8 + (t & 7) + shift) % 40;
    return ((bb * 40 + gr) * 40 + gc) * 180;
  };

  // Phase 1: LayerNorm1 -> XLN (bf16, zero-padded to 192 cols)
  for (int t = wave; t < 64; t += 8) {
    int gb = tokbase(t);
    float s = 0.f, sq = 0.f;
    for (int ch = lane; ch < 180; ch += 32) {
      float v = hin[gb + ch];
      s += v; sq += v * v;
    }
#pragma unroll
    for (int m = 1; m < 32; m <<= 1) {
      s += __shfl_xor(s, m, 32);
      sq += __shfl_xor(sq, m, 32);
    }
    float mean = s * (1.f / 180.f);
    float rstd = rsqrtf(sq * (1.f / 180.f) - mean * mean + 1e-5f);
    for (int ch = lane; ch < 192; ch += 32) {
      float o = (ch < 180) ? ((hin[gb + ch] - mean) * rstd * ln1g[ch] + ln1b[ch]) : 0.f;
      XLN[t * CP + ch] = f2bf(o);
    }
  }
  __syncthreads();

  // Phase 2: QKV GEMM (64x192 @ 192x576); each B fragment feeds 2 WMMAs
  {
    v16bf a0[6], a1[6];
#pragma unroll
    for (int kt = 0; kt < 6; ++kt) {
      a0[kt] = frag_a_lds(XLN + mt0 * 16 * CP + kt * 32, CP);
      a1[kt] = frag_a_lds(XLN + mt1 * 16 * CP + kt * 32, CP);
    }
    for (int nt = ntq; nt < 36; nt += 4) {
      int opad = nt * 16 + nn;
      int seg = opad / 192, cc = opad % 192, hh = cc >> 5, dd = cc & 31;
      float bias = (dd < 30) ? qkvb[seg * 180 + hh * 30 + dd] : 0.f;
      v8f c0 = v8f_fill(bias), c1 = v8f_fill(bias);
      const u16* bt = wqkv + nt * 6 * 512;
      __builtin_prefetch((const void*)(bt + 4 * 6 * 512), 0, 1);
#pragma unroll
      for (int kt = 0; kt < 6; ++kt) {
        v16bf b = frag_b_glob(bt + kt * 512);
        c0 = wmma_bf16(a0[kt], b, c0);
        c1 = wmma_bf16(a1[kt], b, c1);
      }
      float scale = (seg == 0) ? 0.18257418583505536f : 1.f;  // 30^-0.5
      u16* dst = (seg == 0) ? Qb : ((seg == 1) ? Kb : Vb);
#pragma unroll
      for (int v = 0; v < 8; ++v) {
        dst[(mt0 * 16 + v + rowoff) * CP + cc] = f2bf(c0[v] * scale);
        dst[(mt1 * 16 + v + rowoff) * CP + cc] = f2bf(c1[v] * scale);
      }
    }
  }
  __syncthreads();

  // Phase 3: S = Q K^T (+relpos bias +mask), softmax in registers; 1 wave/head
  if (wave < 6) {
    int h = wave;
    v16bf bk[4];
#pragma unroll
    for (int nt = 0; nt < 4; ++nt)
      bk[nt] = frag_b_nmaj(Kb + nt * 16 * CP + h * 32, CP);
    u16* Sh = Sb + h * 64 * SP;
    for (int qt = 0; qt < 4; ++qt) {
      v16bf aq = frag_a_lds(Qb + qt * 16 * CP + h * 32, CP);
      v8f s[4];
#pragma unroll
      for (int nt = 0; nt < 4; ++nt) s[nt] = wmma_bf16(aq, bk[nt], v8f_fill(0.f));
#pragma unroll
      for (int v = 0; v < 8; ++v) {
        int at = qt * 16 + v + rowoff;            // query token in window
        int ai = at >> 3, aj = at & 7;
        int rega = regof(wr * 8 + ai) * 3 + regof(wc * 8 + aj);
        float e[4];
        float mx = -1e30f;
#pragma unroll
        for (int nt = 0; nt < 4; ++nt) {
          int btk = nt * 16 + nn;                 // key token
          int bi = btk >> 3, bj = btk & 7;
          float val = s[nt][v] + rpb[((ai - bi + 7) * 15 + (aj - bj + 7)) * 6 + h];
          if (shift) {
            int regb = regof(wr * 8 + bi) * 3 + regof(wc * 8 + bj);
            if (regb != rega) val -= 100.f;
          }
          e[nt] = val;
          mx = fmaxf(mx, val);
        }
#pragma unroll
        for (int m = 1; m < 16; m <<= 1) mx = fmaxf(mx, __shfl_xor(mx, m, 32));
        float sum = 0.f;
#pragma unroll
        for (int nt = 0; nt < 4; ++nt) {
          e[nt] = __expf(e[nt] - mx);
          sum += e[nt];
        }
#pragma unroll
        for (int m = 1; m < 16; m <<= 1) sum += __shfl_xor(sum, m, 32);
        float inv = 1.f / sum;
#pragma unroll
        for (int nt = 0; nt < 4; ++nt)
          Sh[at * SP + nt * 16 + nn] = f2bf(e[nt] * inv);
      }
    }
  }
  __syncthreads();

  // Phase 4: attn @ V  (6 heads x 4x2 tiles x K=64 over all 8 waves)
  for (int job = wave; job < 48; job += 8) {
    int h = job >> 3, rr = job & 7, amt = rr >> 1, ntv = rr & 1;
    v8f c = v8f_fill(0.f);
#pragma unroll
    for (int kt = 0; kt < 2; ++kt) {
      v16bf a = frag_a_lds(Sb + h * 64 * SP + amt * 16 * SP + kt * 32, SP);
      v16bf b = frag_b_kmaj(Vb + kt * 32 * CP + h * 32 + ntv * 16, CP);
      c = wmma_bf16(a, b, c);
    }
#pragma unroll
    for (int v = 0; v < 8; ++v)
      ATT[(amt * 16 + v + rowoff) * CP + h * 32 + ntv * 16 + nn] = f2bf(c[v]);
  }
  __syncthreads();

  // Phase 5: proj GEMM + shortcut residual -> hout (f32) and X2 (LDS)
  {
    v16bf a0[6], a1[6];
#pragma unroll
    for (int kt = 0; kt < 6; ++kt) {
      a0[kt] = frag_a_lds(ATT + mt0 * 16 * CP + kt * 32, CP);
      a1[kt] = frag_a_lds(ATT + mt1 * 16 * CP + kt * 32, CP);
    }
    for (int nt = ntq; nt < 12; nt += 4) {
      int o = nt * 16 + nn;
      float bias = (o < 180) ? projb[o] : 0.f;
      v8f c0 = v8f_fill(bias), c1 = v8f_fill(bias);
#pragma unroll
      for (int kt = 0; kt < 6; ++kt) {
        v16bf b = frag_b_glob(wproj + (nt * 6 + kt) * 512);
        c0 = wmma_bf16(a0[kt], b, c0);
        c1 = wmma_bf16(a1[kt], b, c1);
      }
      if (o < 180) {
#pragma unroll
        for (int v = 0; v < 8; ++v) {
          int r0 = mt0 * 16 + v + rowoff, r1 = mt1 * 16 + v + rowoff;
          int g0 = tokbase(r0), g1 = tokbase(r1);
          float v0 = c0[v] + hin[g0 + o], v1 = c1[v] + hin[g1 + o];
          hout[g0 + o] = v0;
          hout[g1 + o] = v1;
          X2[r0 * 180 + o] = v0;
          X2[r1 * 180 + o] = v1;
        }
      }
    }
  }
  __syncthreads();

  // Phase 6: LayerNorm2 -> XLN (bf16)
  for (int t = wave; t < 64; t += 8) {
    float s = 0.f, sq = 0.f;
    for (int ch = lane; ch < 180; ch += 32) {
      float v = X2[t * 180 + ch];
      s += v; sq += v * v;
    }
#pragma unroll
    for (int m = 1; m < 32; m <<= 1) {
      s += __shfl_xor(s, m, 32);
      sq += __shfl_xor(sq, m, 32);
    }
    float mean = s * (1.f / 180.f);
    float rstd = rsqrtf(sq * (1.f / 180.f) - mean * mean + 1e-5f);
    for (int ch = lane; ch < 192; ch += 32) {
      float o = (ch < 180) ? ((X2[t * 180 + ch] - mean) * rstd * ln2g[ch] + ln2b[ch]) : 0.f;
      XLN[t * CP + ch] = f2bf(o);
    }
  }
  __syncthreads();

  // Phase 7: fc1 GEMM + exact GELU -> H1 (64x736 bf16, padded rows)
  {
    v16bf a0[6], a1[6];
#pragma unroll
    for (int kt = 0; kt < 6; ++kt) {
      a0[kt] = frag_a_lds(XLN + mt0 * 16 * CP + kt * 32, CP);
      a1[kt] = frag_a_lds(XLN + mt1 * 16 * CP + kt * 32, CP);
    }
    for (int nt = ntq; nt < 46; nt += 4) {
      int o = nt * 16 + nn;
      float bias = (o < 720) ? fc1b[o] : 0.f;
      v8f c0 = v8f_fill(bias), c1 = v8f_fill(bias);
      const u16* bt = wfc1 + nt * 6 * 512;
      __builtin_prefetch((const void*)(bt + 4 * 6 * 512), 0, 1);
#pragma unroll
      for (int kt = 0; kt < 6; ++kt) {
        v16bf b = frag_b_glob(bt + kt * 512);
        c0 = wmma_bf16(a0[kt], b, c0);
        c1 = wmma_bf16(a1[kt], b, c1);
      }
#pragma unroll
      for (int v = 0; v < 8; ++v) {
        float x0 = c0[v], x1 = c1[v];
        float g0 = 0.5f * x0 * (1.f + erff(x0 * 0.70710678118654752f));
        float g1 = 0.5f * x1 * (1.f + erff(x1 * 0.70710678118654752f));
        H1[(mt0 * 16 + v + rowoff) * HP + o] = f2bf(g0);
        H1[(mt1 * 16 + v + rowoff) * HP + o] = f2bf(g1);
      }
    }
  }
  __syncthreads();

  // Phase 8: fc2 GEMM (K=736) + add into hout (same thread wrote x2 there)
  {
    v8f acc0[3], acc1[3];
#pragma unroll
    for (int j = 0; j < 3; ++j) {
      int o = (ntq + 4 * j) * 16 + nn;
      float bias = (o < 180) ? fc2b[o] : 0.f;
      acc0[j] = v8f_fill(bias);
      acc1[j] = v8f_fill(bias);
    }
    for (int kt = 0; kt < 23; ++kt) {
      v16bf a0 = frag_a_lds(H1 + mt0 * 16 * HP + kt * 32, HP);
      v16bf a1 = frag_a_lds(H1 + mt1 * 16 * HP + kt * 32, HP);
#pragma unroll
      for (int j = 0; j < 3; ++j) {
        v16bf b = frag_b_glob(wfc2 + ((ntq + 4 * j) * 23 + kt) * 512);
        acc0[j] = wmma_bf16(a0, b, acc0[j]);
        acc1[j] = wmma_bf16(a1, b, acc1[j]);
      }
    }
#pragma unroll
    for (int j = 0; j < 3; ++j) {
      int o = (ntq + 4 * j) * 16 + nn;
      if (o < 180) {
#pragma unroll
        for (int v = 0; v < 8; ++v) {
          int r0 = mt0 * 16 + v + rowoff, r1 = mt1 * 16 + v + rowoff;
          hout[tokbase(r0) + o] += acc0[j][v];
          hout[tokbase(r1) + o] += acc1[j][v];
        }
      }
    }
  }
}

// ---------------- 3x3 conv (9 tap-GEMMs) + residual, NCHW out ----------------
__global__ __launch_bounds__(256, 1) void conv_res_kernel(
    const float* __restrict__ h2, const float* __restrict__ xin,
    const u16* __restrict__ wconv, const float* __restrict__ convb,
    float* __restrict__ out) {
  __shared__ u16 TILE[64 * CP];  // 64 pixels x 192 bf16 (padded rows)
  int bidx = blockIdx.x;
  int bb = bidx / 25, tid = bidx % 25, tr = tid / 5, tc = tid % 5;
  int wave = threadIdx.x >> 5, lane = threadIdx.x & 31;
  int nn = lane & 15, rowoff = (lane >> 4) << 3;
  int mtp = wave & 1, ntq = wave >> 1;
  int mt0 = mtp * 2, mt1 = mt0 + 1;
  v8f acc0[3], acc1[3];
#pragma unroll
  for (int j = 0; j < 3; ++j) {
    int o = (ntq + 4 * j) * 16 + nn;
    float bias = (o < 180) ? convb[o] : 0.f;
    acc0[j] = v8f_fill(bias);
    acc1[j] = v8f_fill(bias);
  }
  for (int tap = 0; tap < 9; ++tap) {
    int dy = tap / 3 - 1, dx = tap % 3 - 1;
    for (int t = wave; t < 64; t += 8) {
      int r = tr * 8 + (t >> 3) + dy, c = tc * 8 + (t & 7) + dx;
      bool ok = ((unsigned)r < 40u) && ((unsigned)c < 40u);
      const float* src = h2 + ((bb * 40 + r) * 40 + c) * 180;
      for (int ch = lane; ch < 192; ch += 32)
        TILE[t * CP + ch] = (ok && ch < 180) ? f2bf(src[ch]) : (u16)0;
    }
    __syncthreads();
    for (int kt = 0; kt < 6; ++kt) {
      v16bf a0 = frag_a_lds(TILE + mt0 * 16 * CP + kt * 32, CP);
      v16bf a1 = frag_a_lds(TILE + mt1 * 16 * CP + kt * 32, CP);
#pragma unroll
      for (int j = 0; j < 3; ++j) {
        v16bf b = frag_b_glob(wconv + ((tap * 12 + ntq + 4 * j) * 6 + kt) * 512);
        acc0[j] = wmma_bf16(a0, b, acc0[j]);
        acc1[j] = wmma_bf16(a1, b, acc1[j]);
      }
    }
    __syncthreads();
  }
#pragma unroll
  for (int j = 0; j < 3; ++j) {
    int o = (ntq + 4 * j) * 16 + nn;
    if (o < 180) {
#pragma unroll
      for (int v = 0; v < 8; ++v) {
        int p0 = mt0 * 16 + v + rowoff, p1 = mt1 * 16 + v + rowoff;
        int r0 = tr * 8 + (p0 >> 3), c0 = tc * 8 + (p0 & 7);
        int r1 = tr * 8 + (p1 >> 3), c1 = tc * 8 + (p1 & 7);
        int i0 = ((bb * 180 + o) * 40 + r0) * 40 + c0;
        int i1 = ((bb * 180 + o) * 40 + r1) * 40 + c1;
        out[i0] = acc0[j][v] + xin[i0];
        out[i1] = acc1[j][v] + xin[i1];
      }
    }
  }
}

extern "C" void kernel_launch(void* const* d_in, const int* in_sizes, int n_in,
                              void* d_out, int out_size, void* d_ws, size_t ws_size,
                              hipStream_t stream) {
  (void)in_sizes; (void)n_in; (void)out_size; (void)ws_size;
  const float* x = (const float*)d_in[0];

  char* ws = (char*)d_ws;
  const size_t SH = 73728000ULL;  // 64*40*40*180 f32 bytes
  float* H0 = (float*)ws;
  float* H1 = (float*)(ws + SH);
  float* H2 = (float*)(ws + 2 * SH);
  u16* WB = (u16*)(ws + 3 * SH);
  // packed-weight sub-offsets (u16 elements)
  u16* wqkv1 = WB + 0;        // 36*6*512  = 110592
  u16* wproj1 = WB + 110592;  // 12*6*512  =  36864
  u16* wfc1a = WB + 147456;   // 46*6*512  = 141312
  u16* wfc2a = WB + 288768;   // 12*23*512 = 141312
  u16* wqkv2 = WB + 430080;
  u16* wproj2 = WB + 540672;
  u16* wfc1b = WB + 577536;
  u16* wfc2b = WB + 718848;
  u16* wconv = WB + 860160;   // 9*12*6*512 = 331776

  // NCHW -> NHWC
  nchw_to_nhwc_kernel<<<72000, 256, 0, stream>>>(x, H0);

  auto prep = [&](u16* dst, const float* src, int ostr, int istr, int soff,
                  int outr, int inr, int NT, int KT, int qkvmode) {
    int total = NT * KT * 512;
    prep_weight_kernel<<<(total + 255) / 256, 256, 0, stream>>>(
        dst, src, ostr, istr, soff, outr, inr, KT, qkvmode, total);
  };
  // stage 1 weights
  prep(wqkv1, (const float*)d_in[3], 180, 1, 0, 540, 180, 36, 6, 1);
  prep(wproj1, (const float*)d_in[5], 180, 1, 0, 180, 180, 12, 6, 0);
  prep(wfc1a, (const float*)d_in[10], 180, 1, 0, 720, 180, 46, 6, 0);
  prep(wfc2a, (const float*)d_in[12], 720, 1, 0, 180, 720, 12, 23, 0);
  // stage 2 weights
  prep(wqkv2, (const float*)d_in[16], 180, 1, 0, 540, 180, 36, 6, 1);
  prep(wproj2, (const float*)d_in[18], 180, 1, 0, 180, 180, 12, 6, 0);
  prep(wfc1b, (const float*)d_in[23], 180, 1, 0, 720, 180, 46, 6, 0);
  prep(wfc2b, (const float*)d_in[25], 720, 1, 0, 180, 720, 12, 23, 0);
  // conv taps: w[o][i][ky][kx] -> 9 GEMM B-matrices
  for (int t = 0; t < 9; ++t)
    prep(wconv + t * 36864, (const float*)d_in[27], 1620, 9, t, 180, 180, 12, 6, 0);

  // Swin block 1 (no shift)
  swin_block_kernel<<<1600, 256, 0, stream>>>(
      H0, H1, wqkv1, wproj1, wfc1a, wfc2a,
      (const float*)d_in[1], (const float*)d_in[2], (const float*)d_in[4],
      (const float*)d_in[6], (const float*)d_in[7], (const float*)d_in[8],
      (const float*)d_in[9], (const float*)d_in[11], (const float*)d_in[13], 0);
  // Swin block 2 (shift = 4)
  swin_block_kernel<<<1600, 256, 0, stream>>>(
      H1, H2, wqkv2, wproj2, wfc1b, wfc2b,
      (const float*)d_in[14], (const float*)d_in[15], (const float*)d_in[17],
      (const float*)d_in[19], (const float*)d_in[20], (const float*)d_in[21],
      (const float*)d_in[22], (const float*)d_in[24], (const float*)d_in[26], 4);
  // conv + residual -> NCHW output
  conv_res_kernel<<<1600, 256, 0, stream>>>(H2, x, wconv,
                                            (const float*)d_in[28],
                                            (float*)d_out);
}
